// simple3DlossOLD_15040975470953
// MI455X (gfx1250) — matrix operand order repaired and checked by reading, not compiled, not run on hardware
//
#include <hip/hip_runtime.h>

// Problem constants from the reference
#define TIMESTEPS 1000
#define BATCH     64
#define HEIGHT    256
#define WIDTH     256
#define NROWS     (BATCH * HEIGHT)   // 16384 (b,x) rows
#define BINS_PAD  1024               // 1000 bins padded to 1024 (pad diffs are 0)

typedef __attribute__((ext_vector_type(16))) _Float16 v16h;
typedef __attribute__((ext_vector_type(8)))  float    v8f;

// ---- CDNA5 async global->LDS path (probe-confirmed present; signature wants
// addrspace(1) int* src, addrspace(3) int* dst, imm offset, imm cpol) ----
#if defined(__has_builtin)
# if __has_builtin(__builtin_amdgcn_global_load_async_to_lds_b32) && \
     __has_builtin(__builtin_amdgcn_s_wait_asynccnt)
#  define USE_ASYNC_LDS 1
# endif
#endif
#ifndef USE_ASYNC_LDS
# define USE_ASYNC_LDS 0
#endif

#if USE_ASYNC_LDS
typedef __attribute__((address_space(1))) int g1_int;   // global (AS1) b32
typedef __attribute__((address_space(3))) int l3_int;   // LDS (AS3) b32
#endif

// One workgroup (8 wave32) per (b,x) row. Holograms live entirely in LDS.
__global__ __launch_bounds__(256)
void holo_row_kernel(const float* __restrict__ rec,
                     const float* __restrict__ tgt,
                     float* __restrict__ partials)
{
    __shared__ int   s_rec[BINS_PAD];      // scatter-max bins (int y, exact)
    __shared__ int   s_tgt[BINS_PAD];
    __shared__ float s_d[BINS_PAD];        // bin differences for WMMA stage
    __shared__ float s_stage[2 * WIDTH];   // async-staged input rows
    __shared__ float s_wsum[2];

    const int tid = threadIdx.x;           // == y column index, 0..255
    const int row = blockIdx.x;            // flat (b,x)
    const float* recRow = rec + (size_t)row * WIDTH;
    const float* tgtRow = tgt + (size_t)row * WIDTH;

#if USE_ASYNC_LDS
    // Kick off async global->LDS row staging; overlaps with bin zeroing below.
    __builtin_amdgcn_global_load_async_to_lds_b32(
        (g1_int*)(recRow + tid), (l3_int*)&s_stage[tid], 0, 0);
    __builtin_amdgcn_global_load_async_to_lds_b32(
        (g1_int*)(tgtRow + tid), (l3_int*)&s_stage[WIDTH + tid], 0, 0);
#else
    const float vr0 = recRow[tid];
    const float vt0 = tgtRow[tid];
#endif

    // Zero both histograms (4 bins per thread) while loads are in flight.
    #pragma unroll
    for (int i = tid; i < BINS_PAD; i += 256) { s_rec[i] = 0; s_tgt[i] = 0; }
    __syncthreads();

#if USE_ASYNC_LDS
    __builtin_amdgcn_s_wait_asynccnt(0);   // each thread reads only its own slot
    const float vr = s_stage[tid];
    const float vt = s_stage[WIDTH + tid];
#else
    const float vr = vr0;
    const float vt = vt0;
#endif

    // Scatter: last-write-wins over increasing y == max-y -> integer ds_max
    // (order-independent => deterministic). t mirrors int(v*1000)-1 with wrap.
    if (vr != 0.0f) {
        int t = (int)(vr * (float)TIMESTEPS) - 1;
        if (t < 0) t += TIMESTEPS;
        atomicMax(&s_rec[t], tid);
    }
    if (vt != 0.0f) {
        int t = (int)(vt * (float)TIMESTEPS) - 1;
        if (t < 0) t += TIMESTEPS;
        atomicMax(&s_tgt[t], tid);
    }
    __syncthreads();

    // Bin differences (exact small integers, f16-representable: |d| <= 255).
    #pragma unroll
    for (int i = tid; i < BINS_PAD; i += 256)
        s_d[i] = (float)(s_rec[i] - s_tgt[i]);
    __syncthreads();

    // Sum of squares via WMMA Gram diagonal: A[m][k] = d[32m+k], B[k][n] = d[32n+k]
    // => D[m][m] = sum_k d[32m+k]^2. One 16x16x32 WMMA covers 512 bins;
    // waves 0 and 1 cover the padded 1024 bins. EXEC is all-ones per wave.
    const int lane = tid & 31;
    const int wid  = tid >> 5;
    if (wid < 2) {
        const int base = wid * 512;
        const int hi = lane >> 4;          // lane group
        const int lo = lane & 15;          // M for A, N for B
        v16h a, b;
        #pragma unroll
        for (int j = 0; j < 16; ++j) {
            // A 16-bit 16x32 layout (ISA 7.12.2): VGPR v, half h, lane-group hi
            //   k = (v>=4 ? 16 : 0) + hi*8 + (v&3)*2 + h
            const int vreg = j >> 1, h = j & 1;
            const int ka = ((vreg >= 4) ? 16 : 0) + hi * 8 + (vreg & 3) * 2 + h;
            a[j] = (_Float16)s_d[base + lo * 32 + ka];
            // B 16-bit 32x16 layout: same K order as A (K = ka) ; N = lane&15
            const int kb = ka;
            b[j] = (_Float16)s_d[base + lo * 32 + kb];
        }
        v8f c = {};
        c = __builtin_amdgcn_wmma_f32_16x16x32_f16(
                /*neg_a=*/false, a, /*neg_b=*/false, b,
                /*c_mod=*/(short)0, c, /*reuse_a=*/false, /*reuse_b=*/false);

        // Extract diagonal: D[m][n] with m = r + 8*hi (VGPR r), n = lo.
        float v = 0.0f;
        #pragma unroll
        for (int r = 0; r < 8; ++r) {
            if (lo == r + hi * 8) v = c[r];
        }
        // Wave32 reduce (deterministic fixed tree).
        #pragma unroll
        for (int off = 16; off > 0; off >>= 1)
            v += __shfl_down(v, off, 32);
        if (lane == 0) s_wsum[wid] = v;
    }
    __syncthreads();
    if (tid == 0) partials[row] = s_wsum[0] + s_wsum[1];
}

// Deterministic fixed-order final reduction of 16384 per-row partials.
__global__ __launch_bounds__(256)
void holo_reduce_kernel(const float* __restrict__ partials, float* __restrict__ out)
{
    __shared__ double sdata[256];
    double acc = 0.0;
    for (int i = threadIdx.x; i < NROWS; i += 256)
        acc += (double)partials[i];
    sdata[threadIdx.x] = acc;
    __syncthreads();
    #pragma unroll
    for (int s = 128; s > 0; s >>= 1) {
        if (threadIdx.x < s) sdata[threadIdx.x] += sdata[threadIdx.x + s];
        __syncthreads();
    }
    if (threadIdx.x == 0)
        out[0] = (float)(sdata[0] / (double)(BATCH * HEIGHT * TIMESTEPS));
}

extern "C" void kernel_launch(void* const* d_in, const int* in_sizes, int n_in,
                              void* d_out, int out_size, void* d_ws, size_t ws_size,
                              hipStream_t stream)
{
    (void)in_sizes; (void)n_in; (void)out_size; (void)ws_size;
    const float* rec = (const float*)d_in[0];   // reconstructed_image
    const float* tgt = (const float*)d_in[1];   // target_image
    float* partials  = (float*)d_ws;            // 16384 floats = 64 KB scratch

    holo_row_kernel<<<NROWS, 256, 0, stream>>>(rec, tgt, partials);
    holo_reduce_kernel<<<1, 256, 0, stream>>>(partials, (float*)d_out);
}
